// MiddleFusionModule_48189533061148
// MI455X (gfx1250) — compile-verified, matrix operand order checked by reading, not run on hardware
//
#include <hip/hip_runtime.h>
#include <hip/hip_bf16.h>
#include <math.h>

typedef __attribute__((ext_vector_type(16))) __bf16 v16bf;
typedef __attribute__((ext_vector_type(8)))  float  v8f;
typedef __attribute__((ext_vector_type(4)))  unsigned int u32x4;
typedef __attribute__((ext_vector_type(8)))  int i32x8;
typedef __attribute__((ext_vector_type(4)))  int i32x4;

#define NODE_DIM   256
#define TEXT_DIM   768
#define HIDDEN_DIM 1024
#define BATCH      64
#define M_TILE     128
#define LDS_PITCH  260   // floats; 256 + 4 pad DWORDs (matches TDM pad cfg)
#define LN_EPS     1e-3f

#if defined(__has_builtin)
#if __has_builtin(__builtin_amdgcn_tensor_load_to_lds)
#define HAVE_TDM 1
#endif
#endif

// ---------------- small text-MLP kernels (negligible work) ----------------

__global__ __launch_bounds__(256) void k_hidden(const float* __restrict__ text,
                                                const float* __restrict__ W1,
                                                const float* __restrict__ b1,
                                                float* __restrict__ h) {
  int gid = blockIdx.x * 256 + threadIdx.x;          // 64*1024
  int b = gid >> 10, c = gid & 1023;
  const float* tr = text + (size_t)b * TEXT_DIM;
  float acc = b1[c];
  for (int k = 0; k < TEXT_DIM; ++k) acc = fmaf(tr[k], W1[(size_t)k * HIDDEN_DIM + c], acc);
  h[gid] = fmaxf(acc, 0.0f);
}

__global__ __launch_bounds__(256) void k_tout(const float* __restrict__ h,
                                              const float* __restrict__ W2,
                                              const float* __restrict__ b2,
                                              float* __restrict__ t) {
  int gid = blockIdx.x * 256 + threadIdx.x;          // 64*256
  int b = gid >> 8, c = gid & 255;
  const float* hr = h + (size_t)b * HIDDEN_DIM;
  float acc = b2[c];
  for (int k = 0; k < HIDDEN_DIM; ++k) acc = fmaf(hr[k], W2[(size_t)k * NODE_DIM + c], acc);
  t[gid] = acc;
}

__global__ __launch_bounds__(256) void k_tg(const float* __restrict__ t,
                                            const float* __restrict__ Wg,
                                            const float* __restrict__ bg,
                                            float* __restrict__ tg) {
  int gid = blockIdx.x * 256 + threadIdx.x;          // 64*256
  int b = gid >> 8, c = gid & 255;
  const float* tr = t + (size_t)b * NODE_DIM;
  const float* w  = Wg + (size_t)NODE_DIM * NODE_DIM;   // rows 256..511 of Wg
  float acc = bg[c];
  for (int k = 0; k < NODE_DIM; ++k) acc = fmaf(tr[k], w[(size_t)k * NODE_DIM + c], acc);
  tg[gid] = acc;
}

// Pack Wg[:256] (K=256 x N=256, row-major) into bf16 WMMA B fragments.
// Layout: [nt(16)][kb(8)][lane(32)][j(16)]; lane l covers N = nt*16 + (l&15);
// element j covers K = kb*32 + 8*(l>>4) + (j&7) + (j>=8 ? 16 : 0).
__global__ __launch_bounds__(256) void k_packB(const float* __restrict__ Wg,
                                               __bf16* __restrict__ wB) {
  int gid = blockIdx.x * 256 + threadIdx.x;          // 16*8*32*16 = 65536
  int j  = gid & 15;
  int l  = (gid >> 4) & 31;
  int kb = (gid >> 9) & 7;
  int nt = gid >> 12;
  int k = kb * 32 + 8 * (l >> 4) + (j & 7) + ((j >> 3) * 16);
  int n = nt * 16 + (l & 15);
  wB[gid] = (__bf16)Wg[(size_t)k * NODE_DIM + n];
}

// ---------------- main fused kernel ----------------

__global__ __launch_bounds__(256) void fuse_main(const float* __restrict__ node_feat,
                                                 const int*   __restrict__ seg_ids,
                                                 const float* __restrict__ tmat,
                                                 const float* __restrict__ tgmat,
                                                 const __bf16* __restrict__ wB,
                                                 const float* __restrict__ gamma,
                                                 const float* __restrict__ beta,
                                                 float* __restrict__ out) {
  __shared__ float sNode[M_TILE * LDS_PITCH];
  __shared__ int   sSeg[M_TILE];

  const int tid    = threadIdx.x;
  const int wgRow0 = blockIdx.x * M_TILE;

  if (tid < M_TILE) sSeg[tid] = seg_ids[wgRow0 + tid];

#ifdef HAVE_TDM
  // Phase 0: one TDM descriptor stages the whole 128x256 f32 tile into LDS,
  // inserting 4 pad DWORDs after every 256 DWORDs (-> LDS_PITCH = 260).
  if (tid < 32) {                          // wave 0 issues the tensor DMA
    const uint64_t gaddr = (uint64_t)(uintptr_t)(node_feat + (size_t)wgRow0 * NODE_DIM);
    const unsigned ldsA  = (unsigned)(uintptr_t)&sNode[0];
    // D# group0: count=1 | lds_addr | global_addr[56:0] | type=2
    u32x4 g0 = { 1u, ldsA, (unsigned)gaddr,
                 (unsigned)((gaddr >> 32) & 0x1FFFFFFu) | 0x80000000u };
    // D# group1: data_size=4B, pad_enable, pad_interval=7 (256 dw), pad_amount=3 (4 dw)
    //            tensor_dim0=256, tensor_dim1=128, tile 256x128, stride0=256
    i32x8 g1 = { (int)((2u << 16) | (1u << 20) | (7u << 22) | (3u << 25)),
                 (int)(256u << 16),      // tensor_dim0[15:0] << 16
                 (int)(128u << 16),      // tensor_dim0 hi | tensor_dim1[15:0]<<16
                 (int)(256u << 16),      // tensor_dim1 hi | tile_dim0<<16
                 128,                    // tile_dim1 | tile_dim2<<16
                 256,                    // tensor_dim0_stride[31:0]
                 0, 0 };
    i32x4 gz4 = { 0, 0, 0, 0 };
    i32x8 gz8 = { 0, 0, 0, 0, 0, 0, 0, 0 };
    __builtin_amdgcn_tensor_load_to_lds(g0, g1, gz4, gz4, gz8, 0);
    __builtin_amdgcn_s_wait_tensorcnt(0);
  }
#else
  // Fallback: per-lane async copies into LDS (128 rows x 64 float4 chunks).
  {
    const unsigned ldsBase = (unsigned)(uintptr_t)&sNode[0];
#pragma unroll
    for (int i = 0; i < 32; ++i) {
      int c    = i * 256 + tid;
      int row  = c >> 6;
      int col4 = c & 63;
      unsigned ldsAddr = ldsBase + (unsigned)((row * LDS_PITCH + col4 * 4) * 4);
      unsigned long long gaddr =
          (unsigned long long)(uintptr_t)(node_feat + (size_t)(wgRow0 + row) * NODE_DIM + col4 * 4);
      asm volatile("global_load_async_to_lds_b128 %0, %1, off"
                   :: "v"(ldsAddr), "v"(gaddr) : "memory");
    }
    asm volatile("s_wait_asynccnt 0x0" ::: "memory");
  }
#endif
  __syncthreads();

  const int wave = tid >> 5;
  const int lane = tid & 31;
  const int half = lane >> 4;     // 0: lanes 0-15, 1: lanes 16-31
  const int lm   = lane & 15;
  const int mW   = wave * 16;     // wave's first local row

  v8f acc[16];
#pragma unroll
  for (int nt = 0; nt < 16; ++nt) acc[nt] = (v8f){0.f,0.f,0.f,0.f,0.f,0.f,0.f,0.f};

  // Phase 1: GEMM — K in 8 blocks of 32; wave computes 16 rows x 256 cols.
#pragma unroll
  for (int kb = 0; kb < 8; ++kb) {
    // A fragment: row mW+lm; K = kb*32 + 8*half + {0..7} and +16.
    const float* src = &sNode[(mW + lm) * LDS_PITCH + kb * 32 + half * 8];
    v16bf a;
#pragma unroll
    for (int j = 0; j < 8; ++j) {
      a[j]     = (__bf16)src[j];
      a[j + 8] = (__bf16)src[j + 16];
    }
    const __bf16* bptr = wB + ((size_t)kb * 32 + lane) * 16;
    v16bf bcur = *(const v16bf*)bptr;            // B fragment, double-buffered
#pragma unroll
    for (int nt = 0; nt < 16; ++nt) {
      v16bf bnext = bcur;
      if (nt < 15) bnext = *(const v16bf*)(bptr + (size_t)(nt + 1) * (8 * 32 * 16));
      acc[nt] = __builtin_amdgcn_wmma_f32_16x16x32_bf16(false, a, false, bcur,
                                                        (short)0, acc[nt],
                                                        false, false);
      bcur = bnext;
    }
  }

  // Phase 2: fused epilogue. C layout: VGPR v holds row v (lanes 0-15) and
  // row v+8 (lanes 16-31), column nt*16 + (lane&15).
#pragma unroll
  for (int v = 0; v < 8; ++v) {
    const int mLoc  = mW + v + 8 * half;
    const int mGlob = wgRow0 + mLoc;
    const int s     = sSeg[mLoc];
    const float* trow  = tmat  + (size_t)s * NODE_DIM;
    const float* tgrow = tgmat + (size_t)s * NODE_DIM;

    float sum = 0.f, sumsq = 0.f;
    float enh[16];
#pragma unroll
    for (int nt = 0; nt < 16; ++nt) {
      const int n = nt * 16 + lm;
      float gp = acc[nt][v] + tgrow[n];
      float g  = 1.0f / (1.0f + __expf(-gp));
      float e  = sNode[mLoc * LDS_PITCH + n] + g * trow[n];
      enh[nt] = e;
      sum   += e;
      sumsq += e * e;
    }
    // Reduce across the 16 lanes holding this row (masks 1..8 stay in-half).
#pragma unroll
    for (int m = 1; m < 16; m <<= 1) {
      sum   += __shfl_xor(sum,   m, 32);
      sumsq += __shfl_xor(sumsq, m, 32);
    }
    const float mean = sum * (1.0f / NODE_DIM);
    const float var  = sumsq * (1.0f / NODE_DIM) - mean * mean;
    const float rstd = rsqrtf(var + LN_EPS);

    float* orow = out + (size_t)mGlob * NODE_DIM;
#pragma unroll
    for (int nt = 0; nt < 16; ++nt) {
      const int n = nt * 16 + lm;
      orow[n] = (enh[nt] - mean) * rstd * gamma[n] + beta[n];
    }
  }
}

// ---------------- launcher ----------------

extern "C" void kernel_launch(void* const* d_in, const int* in_sizes, int n_in,
                              void* d_out, int out_size, void* d_ws, size_t ws_size,
                              hipStream_t stream) {
  const float* node_feat = (const float*)d_in[0];
  const float* text_feat = (const float*)d_in[1];
  const int*   seg_ids   = (const int*)  d_in[2];
  const float* W1        = (const float*)d_in[3];
  const float* b1        = (const float*)d_in[4];
  const float* W2        = (const float*)d_in[5];
  const float* b2        = (const float*)d_in[6];
  const float* Wg        = (const float*)d_in[7];
  const float* bg        = (const float*)d_in[8];
  const float* gamma     = (const float*)d_in[9];
  const float* beta      = (const float*)d_in[10];
  float* out = (float*)d_out;

  char* ws = (char*)d_ws;
  float*  h  = (float*)(ws);                       // 64*1024 f32 = 256 KB
  float*  t  = (float*)(ws + (256 << 10));         // 64*256 f32  =  64 KB
  float*  tg = (float*)(ws + (320 << 10));         // 64*256 f32  =  64 KB
  __bf16* wB = (__bf16*)(ws + (384 << 10));        // 65536 bf16  = 128 KB

  const int total_nodes = in_sizes[0] / NODE_DIM;  // 131072

  k_hidden<<<(BATCH * HIDDEN_DIM) / 256, 256, 0, stream>>>(text_feat, W1, b1, h);
  k_tout  <<<(BATCH * NODE_DIM) / 256,   256, 0, stream>>>(h, W2, b2, t);
  k_tg    <<<(BATCH * NODE_DIM) / 256,   256, 0, stream>>>(t, Wg, bg, tg);
  k_packB <<<65536 / 256,                256, 0, stream>>>(Wg, wB);
  fuse_main<<<total_nodes / M_TILE, 256, 0, stream>>>(node_feat, seg_ids, t, tg,
                                                      wB, gamma, beta, out);
}
